// USES2_Swin_66211215835613
// MI455X (gfx1250) — matrix operand order repaired and checked by reading, not compiled, not run on hardware
//
#include <hip/hip_runtime.h>
#include <hip/hip_bf16.h>
#include <stdint.h>

// ---------------------------------------------------------------------------
// USES2 Swin for MI455X (gfx1250, wave32).  All GEMMs via V_WMMA_F32_16X16X32_F16
// with f32 accumulation; fused epilogues; attention fully in LDS per (window,head).
// Fragment staging is fully vectorized: A-type fragments = 2x b128 LDS loads,
// B-type fragments = fragment-major weight layout -> 1x 32B LDS load.
// ---------------------------------------------------------------------------

typedef _Float16 f16;
typedef __attribute__((ext_vector_type(16))) _Float16 v16h;
typedef __attribute__((ext_vector_type(8)))  _Float16 v8h;
typedef __attribute__((ext_vector_type(8)))  float    v8f;

#define BQ 2
#define FQ 130
#define TQ 192
#define LQ (FQ*TQ)            // 24960
#define NWIN 312              // 13*24 windows per image
#define M0 99840              // 4 * LQ (block0 token rows)
#define M1 49920              // 2 * LQ (block1 token rows)
#define EPSV 1e-5f

__device__ inline v16h cat8(v8h lo, v8h hi) {
    return __builtin_shufflevector(lo, hi, 0, 1, 2, 3, 4, 5, 6, 7, 8, 9, 10, 11, 12, 13, 14, 15);
}

// -------------------------- WMMA GEMM (64x64x32 tiles) ----------------------

enum { E_F16 = 0, E_F32, E_QKV, E_PROJ, E_GELU, E_RES, E_RELU, E_PRELU, E_OUT };

struct GemmArgs {
    const f16* A;        // M x K row-major (f16)
    const f16* Bt;       // fragment-major weight layout (see k_conv_w_t)
    const float* bias;   // length N or nullptr
    int M, N, K, epi;
    float* outf;
    f16*   outh;
    float* res;          // residual target
    const float* scal;   // PReLU alpha (device scalar)
    f16 *Qb, *Kb, *Vb;   // qkv split targets
    int shifted;
    int Ldim;
};

__device__ inline void epi_store(const GemmArgs& g, int m, int n, float v) {
    if (g.bias) v += g.bias[n];
    switch (g.epi) {
    case E_F16:
        g.outh[(size_t)m * g.N + n] = (f16)v;
        break;
    case E_F32:
        g.outf[(size_t)m * g.N + n] = v;
        break;
    case E_QKV: {
        int which = n >> 6, head = (n >> 4) & 3, e = n & 15;
        int bw = m / 80, t = m - bw * 80;
        size_t idx = (((size_t)bw * 4 + head) * 80 + t) * 16 + e;
        if (which == 0)      g.Qb[idx] = (f16)(v * 0.25f);   // hd^-0.5
        else if (which == 1) g.Kb[idx] = (f16)v;
        else                 g.Vb[idx] = (f16)v;
    } break;
    case E_PROJ: {   // window-reverse (+unshift) scatter-add into residual stream
        int bw = m / 80, t = m - bw * 80;
        int bp = bw / NWIN, wi = bw - bp * NWIN;
        int wf = wi / 24, wt = wi - wf * 24;
        int i = t >> 3, j = t & 7;
        int h = wf * 10 + i, w = wt * 8 + j;
        if (g.shifted) { h += 5; if (h >= FQ) h -= FQ; w += 4; if (w >= TQ) w -= TQ; }
        size_t l = (size_t)h * TQ + w;
        g.res[(((size_t)bp * LQ) + l) * 64 + n] += v;
    } break;
    case E_GELU: {   // exact gelu
        float u = 0.5f * v * (1.0f + erff(v * 0.70710678118f));
        g.outh[(size_t)m * g.N + n] = (f16)u;
    } break;
    case E_RES:
        g.res[(size_t)m * 64 + n] += v;
        break;
    case E_RELU:
        g.outf[(size_t)m * g.N + n] = fmaxf(v, 0.f);
        break;
    case E_PRELU: {
        float a = *g.scal;
        g.outf[(size_t)m * g.N + n] = (v >= 0.f) ? v : a * v;
    } break;
    case E_OUT: {    // (B, 256, L) layout of d_out
        int b = m / g.Ldim, l = m - b * g.Ldim;
        g.outf[(((size_t)b * 256) + n) * g.Ldim + l] = v;
    } break;
    }
}

__global__ __launch_bounds__(256) void gemm_f16_kernel(GemmArgs g) {
    __shared__ __align__(16) f16 As[64][40];    // row-major A tile (padded rows)
    __shared__ __align__(32) f16 BsF[2048];     // fragment-major B tile: 4 ntiles x 32 slots x 16
    const int tid  = threadIdx.x;
    const int wave = tid >> 5, lane = tid & 31;
    const int half = lane >> 4, l15 = lane & 15;
    const int wm = wave >> 1, wn = wave & 1;           // 4x2 wave grid
    const int bm = blockIdx.x * 64, bn = blockIdx.y * 64;
    v8f acc0 = {}, acc1 = {};
    for (int k0 = 0; k0 < g.K; k0 += 32) {
        {
            int r = tid >> 2, cc = (tid & 3) * 8;
            const f16* src = g.A + (size_t)(bm + r) * g.K + k0 + cc;
            *(float4*)&As[r][cc] = *(const float4*)src;
        }
        {   // contiguous 4KB copy: ntiles bn/16 .. bn/16+3 of k-block k0/32
            const f16* src = g.Bt + (((size_t)(k0 >> 5) * (g.N >> 4) + (bn >> 4)) * 32) * 16;
            *(float4*)&BsF[tid * 8] = *(const float4*)(src + tid * 8);
        }
        __syncthreads();
        const int mrow = wm * 16 + l15;
        v8h alo = *(const v8h*)&As[mrow][half * 16];
        v8h ahi = *(const v8h*)&As[mrow][half * 16 + 8];
        v16h a  = cat8(alo, ahi);
        v16h b0 = *(const v16h*)&BsF[(((wn * 2 + 0) * 32) + lane) * 16];
        v16h b1 = *(const v16h*)&BsF[(((wn * 2 + 1) * 32) + lane) * 16];
        acc0 = __builtin_amdgcn_wmma_f32_16x16x32_f16(false, a, false, b0, (short)0, acc0, false, false);
        acc1 = __builtin_amdgcn_wmma_f32_16x16x32_f16(false, a, false, b1, (short)0, acc1, false, false);
        __syncthreads();
    }
    int mb = bm + wm * 16 + half * 8;
    int n0 = bn + wn * 32 + l15, n1 = n0 + 16;
#pragma unroll
    for (int r = 0; r < 8; ++r) {
        epi_store(g, mb + r, n0, acc0[r]);
        epi_store(g, mb + r, n1, acc1[r]);
    }
}

// ----------------------------- attention kernel -----------------------------

__device__ inline int maskreg1(int p, int lim, int wsz) {
    return (p < lim - wsz) ? 0 : ((p < lim - (wsz >> 1)) ? 1 : 2);
}

__global__ __launch_bounds__(256) void attn_kernel(const f16* Qb, const f16* Kb, const f16* Vb,
                                                   const float* rpb, f16* Attn, int shifted) {
    __shared__ __align__(16) f16  Qs[80][32];     // row-major, k 16..31 zero
    __shared__ __align__(32) f16  KsF[5 * 32 * 16]; // fragment-major K^T (upper-k half zero)
    __shared__ __align__(32) f16  VsF[3 * 32 * 16]; // fragment-major V (k rows 80..95 zero)
    __shared__ float Ss[80][84];
    __shared__ __align__(16) f16  Ps[80][96];
    const int bh = blockIdx.x;
    const int bw = bh >> 2, head = bh & 3;
    const int wi = bw % NWIN;
    const int wf = wi / 24, wt = wi - wf * 24;
    const int tid = threadIdx.x;
    const size_t base = (size_t)bh * 1280;   // 80*16
    float4 fz; fz.x = 0.f; fz.y = 0.f; fz.z = 0.f; fz.w = 0.f;
    for (int i = tid; i < 160; i += 256) {   // Q + K fills, 16B chunks
        int r = i >> 1, cg = (i & 1) * 8;
        *(float4*)&Qs[r][cg]      = *(const float4*)(Qb + base + r * 16 + cg);
        *(float4*)&Qs[r][16 + cg] = fz;
        int slot = (((r >> 4) * 32) + (r & 15)) * 16;      // half=0 slot
        *(float4*)&KsF[slot + cg]       = *(const float4*)(Kb + base + r * 16 + cg);
        *(float4*)&KsF[slot + 256 + cg] = fz;              // half=1 slot (k 16..31) zero
    }
    for (int i = tid; i < 96 * 16; i += 256) {  // V transpose fill (scalar stores, once)
        int vrow = i >> 4, n = i & 15;
        f16 val = (vrow < 80) ? Vb[base + vrow * 16 + n] : (f16)0.0f;
        int kt = vrow >> 5, kh = (vrow >> 4) & 1, e = vrow & 15;
        VsF[((kt * 32 + kh * 16 + n) * 16) + e] = val;
    }
    __syncthreads();
    const int wave = tid >> 5, lane = tid & 31, half = lane >> 4, l15 = lane & 15;
    // S = Q @ K^T  (K=16 zero-padded to 32), 25 16x16 tiles over 8 waves
    for (int t5 = wave; t5 < 25; t5 += 8) {
        int ti = t5 / 5, tj = t5 - ti * 5;
        v8h alo = *(const v8h*)&Qs[ti * 16 + l15][half * 16];
        v8h ahi = *(const v8h*)&Qs[ti * 16 + l15][half * 16 + 8];
        v16h a  = cat8(alo, ahi);
        v16h b  = *(const v16h*)&KsF[((tj * 32) + lane) * 16];
        v8f acc = {};
        acc = __builtin_amdgcn_wmma_f32_16x16x32_f16(false, a, false, b, (short)0, acc, false, false);
        int n = tj * 16 + l15;
        int fj = n >> 3, wjv = n & 7;
#pragma unroll
        for (int r = 0; r < 8; ++r) {
            int m = ti * 16 + half * 8 + r;
            int fi = m >> 3, wiv = m & 7;
            float v = acc[r] + rpb[((fi - fj + 9) * 15 + (wiv - wjv + 7)) * 4 + head];
            if (shifted) {
                int ri = maskreg1(wf * 10 + fi, FQ, 10) * 3 + maskreg1(wt * 8 + wiv, TQ, 8);
                int rj = maskreg1(wf * 10 + fj, FQ, 10) * 3 + maskreg1(wt * 8 + wjv, TQ, 8);
                if (ri != rj) v -= 100.f;
            }
            Ss[m][n] = v;
        }
    }
    __syncthreads();
    if (tid < 80) {    // row softmax, pack P (80x96, zero-pad cols) as f16
        float mx = -1e30f;
        for (int j2 = 0; j2 < 80; ++j2) mx = fmaxf(mx, Ss[tid][j2]);
        float s = 0.f;
        for (int j2 = 0; j2 < 80; ++j2) { float e = __expf(Ss[tid][j2] - mx); Ss[tid][j2] = e; s += e; }
        float inv = 1.f / s;
        for (int j2 = 0; j2 < 96; ++j2) Ps[tid][j2] = (j2 < 80) ? (f16)(Ss[tid][j2] * inv) : (f16)0.0f;
    }
    __syncthreads();
    if (wave < 5) {    // O = P @ V, K padded 80->96, one 16x16 tile per wave
        v8f acc = {};
#pragma unroll
        for (int kc = 0; kc < 96; kc += 32) {
            v8h alo = *(const v8h*)&Ps[wave * 16 + l15][kc + half * 16];
            v8h ahi = *(const v8h*)&Ps[wave * 16 + l15][kc + half * 16 + 8];
            v16h a  = cat8(alo, ahi);
            v16h b  = *(const v16h*)&VsF[(((kc >> 5) * 32) + lane) * 16];
            acc = __builtin_amdgcn_wmma_f32_16x16x32_f16(false, a, false, b, (short)0, acc, false, false);
        }
#pragma unroll
        for (int r = 0; r < 8; ++r) {
            int m = wave * 16 + half * 8 + r;
            Attn[(((size_t)bw * 80) + m) * 64 + head * 16 + l15] = (f16)acc[r];
        }
    }
}

// ----------------------------- pointwise kernels ----------------------------

// weight -> fragment-major f16 layout:
//   idx(k,n) = ((k/32 * Ntot/16 + n/16) * 32 + ((k>>4)&1)*16 + (n&15)) * 16 + (k&15)
__global__ void k_conv_w_t(const float* W, f16* Bt, int K, int Nw, int Ntot, int nofs) {
    int idx = blockIdx.x * 256 + threadIdx.x;
    if (idx >= K * Nw) return;
    int k = idx / Nw, n = idx - k * Nw;
    int gn = nofs + n;
    int kt = k >> 5, kh = (k >> 4) & 1, e = k & 15;
    int nt = gn >> 4, nr = gn & 15;
    size_t dst = ((((size_t)kt * (Ntot >> 4) + nt) * 32) + kh * 16 + nr) * 16 + e;
    Bt[dst] = (f16)W[(size_t)n * K + k];
}

__global__ void k_copy_f32(float* dst, const float* src, int n) {
    int i = blockIdx.x * blockDim.x + threadIdx.x;
    if (i < n) dst[i] = src[i];
}

// channelwise LN over 256 input features -> f16 token rows
__global__ void k_cln(const float* x, const float* g, const float* b, f16* Xn) {
    int r = blockIdx.x * 256 + threadIdx.x;
    if (r >= M0) return;
    int bc = r / LQ, l = r - bc * LQ;
    const float* src = x + (size_t)bc * 256 * LQ + l;
    float s = 0.f, s2 = 0.f;
    for (int n = 0; n < 256; ++n) { float v = src[(size_t)n * LQ]; s += v; s2 += v * v; }
    float m = s * (1.f / 256.f), var = s2 * (1.f / 256.f) - m * m;
    float inv = rsqrtf(var + EPSV);
    f16* dst = Xn + (size_t)r * 256;
    for (int n = 0; n < 256; ++n) {
        float v = src[(size_t)n * LQ];
        dst[n] = (f16)(g[n] * (v - m) * inv + b[n]);
    }
}

// token LN (64) + shift/window gather -> window-ordered f16 rows
__global__ void k_ln_win(const float* X, const float* g, const float* b, f16* Aw, int shifted, int M) {
    int r = blockIdx.x * 256 + threadIdx.x;
    if (r >= M) return;
    int bw = r / 80, t = r - bw * 80;
    int bp = bw / NWIN, wi = bw - bp * NWIN;
    int wf = wi / 24, wt = wi - wf * 24;
    int i = t >> 3, j = t & 7;
    int h = wf * 10 + i, w = wt * 8 + j;
    if (shifted) { h += 5; if (h >= FQ) h -= FQ; w += 4; if (w >= TQ) w -= TQ; }
    const float* src = X + (((size_t)bp * LQ) + (size_t)h * TQ + w) * 64;
    float s = 0.f, s2 = 0.f;
    for (int n = 0; n < 64; ++n) { float v = src[n]; s += v; s2 += v * v; }
    float m = s * (1.f / 64.f), var = s2 * (1.f / 64.f) - m * m, inv = rsqrtf(var + EPSV);
    f16* dst = Aw + (size_t)r * 64;
    for (int n = 0; n < 64; ++n) dst[n] = (f16)(g[n] * (src[n] - m) * inv + b[n]);
}

__global__ void k_ln_plain(const float* X, const float* g, const float* b, f16* O, int M) {
    int r = blockIdx.x * 256 + threadIdx.x;
    if (r >= M) return;
    const float* src = X + (size_t)r * 64;
    float s = 0.f, s2 = 0.f;
    for (int n = 0; n < 64; ++n) { float v = src[n]; s += v; s2 += v * v; }
    float m = s * (1.f / 64.f), var = s2 * (1.f / 64.f) - m * m, inv = rsqrtf(var + EPSV);
    f16* dst = O + (size_t)r * 64;
    for (int n = 0; n < 64; ++n) dst[n] = (f16)(g[n] * (src[n] - m) * inv + b[n]);
}

__global__ void k_mean_ch(const float* X, float* X1) {
    int idx = blockIdx.x * 256 + threadIdx.x;
    if (idx >= M1 * 64) return;
    int r = idx >> 6, n = idx & 63;
    int b = r / LQ, l = r - b * LQ;
    float v0 = X[(((size_t)(b * 2 + 0) * LQ) + l) * 64 + n];
    float v1 = X[(((size_t)(b * 2 + 1) * LQ) + l) * 64 + n];
    X1[(size_t)idx] = 0.5f * (v0 + v1);
}

__global__ void k_prelu_pack(const float* X1, const float* alpha, f16* A) {
    int idx = blockIdx.x * 256 + threadIdx.x;
    if (idx >= M1 * 64) return;
    float v = X1[(size_t)idx], a = *alpha;
    A[(size_t)idx] = (f16)((v >= 0.f) ? v : a * v);
}

// ------------------------------- TAC kernels --------------------------------

// pack (b,t,c,f) rows x 64 from residual stream (bc, f*T+t, 64)
__global__ void k_pack_tac(const float* X, f16* A) {
    int idx = blockIdx.x * 256 + threadIdx.x;
    if (idx >= M0 * 64) return;
    int rid = idx >> 6, k = idx & 63;
    int f = rid % FQ; int q = rid / FQ; int c = q & 1; int q2 = q >> 1;
    int t = q2 % TQ;  int b = q2 / TQ;
    A[(size_t)idx] = (f16)X[(((size_t)(b * 2 + c) * LQ) + (size_t)f * TQ + t) * 64 + k];
}

// (bc,f,t) rows x 192 gathered from tr-row-major OutTr
__global__ void k_pack_ca(const float* OutTr, f16* A) {
    int idx = blockIdx.x * 256 + threadIdx.x;
    if (idx >= M0 * 192) return;
    int pos = idx / 192, n = idx - pos * 192;
    int bc = pos / LQ, rem = pos - bc * LQ;
    int f = rem / TQ, t = rem - f * TQ;
    int b = bc >> 1, c = bc & 1;
    size_t rid = (((size_t)(b * TQ + t) * 2 + c) * FQ + f);
    A[(size_t)idx] = (f16)OutTr[rid * 192 + n];
}

// per-head chunk LN in place
__global__ void k_chunk_ln(float* Y, const float* g, const float* be, int rows, int N, int cs) {
    int id = blockIdx.x * 256 + threadIdx.x;
    int nch = N / cs;
    if (id >= rows * nch) return;
    int row = id / nch, ch = id - row * nch;
    float* p = Y + (size_t)row * N + ch * cs;
    float s = 0.f, s2 = 0.f;
    for (int e = 0; e < cs; ++e) { float v = p[e]; s += v; s2 += v * v; }
    float m = s / cs, var = s2 / cs - m * m, inv = rsqrtf(var + EPSV);
    const float* gg = g + ch * cs; const float* bb = be + ch * cs;
    for (int e = 0; e < cs; ++e) p[e] = gg[e] * (p[e] - m) * inv + bb[e];
}

// channel-attention logits: 32 blocks, full-length dots
__global__ void k_ca_dots(const float* QW, const float* KW, float* logits) {
    __shared__ float red[256];
    int id = blockIdx.x;                       // (b,h,c,c2)
    int b = id >> 4, h = (id >> 2) & 3, c = (id >> 1) & 1, c2 = id & 1;
    size_t qbase = ((size_t)(b * 2 + c)  * LQ) * 256 + h * 64;
    size_t kbase = ((size_t)(b * 2 + c2) * LQ) * 256 + h * 64;
    float s = 0.f;
    for (int i = threadIdx.x; i < LQ * 64; i += 256) {
        int pos = i >> 6, e = i & 63;
        s += QW[qbase + (size_t)pos * 256 + e] * KW[kbase + (size_t)pos * 256 + e];
    }
    red[threadIdx.x] = s; __syncthreads();
    for (int st = 128; st > 0; st >>= 1) {
        if (threadIdx.x < st) red[threadIdx.x] += red[threadIdx.x + st];
        __syncthreads();
    }
    if (threadIdx.x == 0) logits[id] = red[0] * rsqrtf((float)(64 * LQ));
}

__global__ void k_ca_soft(const float* logits, float* wts) {
    int t = threadIdx.x;
    if (t < 16) {
        float l0 = logits[t * 2], l1 = logits[t * 2 + 1];
        float m = fmaxf(l0, l1);
        float e0 = __expf(l0 - m), e1 = __expf(l1 - m);
        float inv = 1.f / (e0 + e1);
        wts[t * 2] = e0 * inv; wts[t * 2 + 1] = e1 * inv;
    }
}

__global__ void k_ca_mix(const float* VW, const float* wts, f16* Apw) {
    int idx = blockIdx.x * 256 + threadIdx.x;
    if (idx >= M0 * 192) return;
    int pos = idx / 192, n = idx - pos * 192;
    int bc = pos / LQ, rem = pos - bc * LQ;
    int b = bc >> 1;
    int h = n / 48;
    const float* w = wts + ((b * 4 + h) * 2 + (bc & 1)) * 2;
    float v = w[0] * VW[(((size_t)(b * 2 + 0) * LQ) + rem) * 192 + n]
            + w[1] * VW[(((size_t)(b * 2 + 1) * LQ) + rem) * 192 + n];
    Apw[(size_t)idx] = (f16)v;
}

// LN(192) + residual(oa) + prelu  -> OA2 (pos-major)
__global__ void k_ln_res_prelu(const float* P1, const float* OutTr, const float* g, const float* be,
                               const float* alpha, float* OA2) {
    int pos = blockIdx.x * 256 + threadIdx.x;
    if (pos >= M0) return;
    const float* p = P1 + (size_t)pos * 192;
    float s = 0.f, s2 = 0.f;
    for (int n = 0; n < 192; ++n) { float v = p[n]; s += v; s2 += v * v; }
    float m = s / 192.f, var = s2 / 192.f - m * m, inv = rsqrtf(var + EPSV);
    int bc = pos / LQ, rem = pos - bc * LQ;
    int f = rem / TQ, t = rem - f * TQ;
    int b = bc >> 1, c = bc & 1;
    size_t rid = (((size_t)(b * TQ + t) * 2 + c) * FQ + f);
    const float* ot = OutTr + rid * 192;
    float a = *alpha;
    float* q = OA2 + (size_t)pos * 192;
    for (int n = 0; n < 192; ++n) {
        float y = g[n] * (p[n] - m) * inv + be[n] + ot[n];
        q[n] = (y >= 0.f) ? y : a * y;
    }
}

// concat [out_tr | OA2-transposed] -> 384-wide f16 rows
__global__ void k_pack_cc(const float* OutTr, const float* OA2, f16* Acc) {
    int idx = blockIdx.x * 256 + threadIdx.x;
    if (idx >= M0 * 384) return;
    int rid = idx / 384, n = idx - rid * 384;
    if (n < 192) {
        Acc[(size_t)idx] = (f16)OutTr[(size_t)rid * 192 + n];
    } else {
        int f = rid % FQ; int q = rid / FQ; int c = q & 1; int q2 = q >> 1;
        int t = q2 % TQ;  int b = q2 / TQ;
        size_t pos = ((size_t)(b * 2 + c) * LQ) + (size_t)f * TQ + t;
        Acc[(size_t)idx] = (f16)OA2[pos * 192 + (n - 192)];
    }
}

// cc LN(64) + transpose + residual add into stream X
__global__ void k_cc_ln_res(const float* C1, const float* g, const float* be, float* X) {
    int rid = blockIdx.x * 256 + threadIdx.x;
    if (rid >= M0) return;
    const float* p = C1 + (size_t)rid * 64;
    float s = 0.f, s2 = 0.f;
    for (int n = 0; n < 64; ++n) { float v = p[n]; s += v; s2 += v * v; }
    float m = s / 64.f, var = s2 / 64.f - m * m, inv = rsqrtf(var + EPSV);
    int f = rid % FQ; int q = rid / FQ; int c = q & 1; int q2 = q >> 1;
    int t = q2 % TQ;  int b = q2 / TQ;
    float* dst = X + (((size_t)(b * 2 + c) * LQ) + (size_t)f * TQ + t) * 64;
    for (int n = 0; n < 64; ++n) dst[n] += g[n] * (p[n] - m) * inv + be[n];
}

// ------------------------------- host side ----------------------------------

static inline void launch_gemm(hipStream_t s, const f16* A, const f16* Bt, const float* bias,
                               int M, int N, int K, int epi,
                               float* outf, f16* outh, float* res, const float* scal,
                               f16* Qb, f16* Kb, f16* Vb, int shifted, int Ldim) {
    GemmArgs g;
    g.A = A; g.Bt = Bt; g.bias = bias; g.M = M; g.N = N; g.K = K; g.epi = epi;
    g.outf = outf; g.outh = outh; g.res = res; g.scal = scal;
    g.Qb = Qb; g.Kb = Kb; g.Vb = Vb; g.shifted = shifted; g.Ldim = Ldim;
    dim3 grid(M / 64, N / 64);
    gemm_f16_kernel<<<grid, 256, 0, s>>>(g);
}

// swin param offsets within a 13-tensor group (dict insertion order)
enum { SW_N1G = 0, SW_N1B, SW_QKVW, SW_QKVB, SW_RPB, SW_PROJW, SW_PROJB,
       SW_N2G, SW_N2B, SW_FC1W, SW_FC1B, SW_FC2W, SW_FC2B };

static void run_swin(hipStream_t s, void* const* din, int base, int Bp, int shifted,
                     float* X, f16* Aw, f16* Qb, f16* Kb, f16* Vb, f16* Attn, f16* Hln, f16* FC1,
                     f16* wqkv, f16* wproj, f16* wfc1, f16* wfc2) {
    const int M = Bp * LQ;
    k_conv_w_t<<<(64 * 192 + 255) / 256, 256, 0, s>>>((const float*)din[base + SW_QKVW], wqkv, 64, 192, 192, 0);
    k_conv_w_t<<<(64 * 64 + 255) / 256, 256, 0, s>>>((const float*)din[base + SW_PROJW], wproj, 64, 64, 64, 0);
    k_conv_w_t<<<(64 * 256 + 255) / 256, 256, 0, s>>>((const float*)din[base + SW_FC1W], wfc1, 64, 256, 256, 0);
    k_conv_w_t<<<(256 * 64 + 255) / 256, 256, 0, s>>>((const float*)din[base + SW_FC2W], wfc2, 256, 64, 64, 0);

    k_ln_win<<<(M + 255) / 256, 256, 0, s>>>(X, (const float*)din[base + SW_N1G],
                                             (const float*)din[base + SW_N1B], Aw, shifted, M);
    launch_gemm(s, Aw, wqkv, (const float*)din[base + SW_QKVB], M, 192, 64, E_QKV,
                nullptr, nullptr, nullptr, nullptr, Qb, Kb, Vb, 0, 0);
    attn_kernel<<<Bp * NWIN * 4, 256, 0, s>>>(Qb, Kb, Vb, (const float*)din[base + SW_RPB], Attn, shifted);
    launch_gemm(s, Attn, wproj, (const float*)din[base + SW_PROJB], M, 64, 64, E_PROJ,
                nullptr, nullptr, X, nullptr, nullptr, nullptr, nullptr, shifted, 0);
    k_ln_plain<<<(M + 255) / 256, 256, 0, s>>>(X, (const float*)din[base + SW_N2G],
                                               (const float*)din[base + SW_N2B], Hln, M);
    launch_gemm(s, Hln, wfc1, (const float*)din[base + SW_FC1B], M, 256, 64, E_GELU,
                nullptr, FC1, nullptr, nullptr, nullptr, nullptr, nullptr, 0, 0);
    launch_gemm(s, FC1, wfc2, (const float*)din[base + SW_FC2B], M, 64, 256, E_RES,
                nullptr, nullptr, X, nullptr, nullptr, nullptr, nullptr, 0, 0);
}

extern "C" void kernel_launch(void* const* d_in, const int* in_sizes, int n_in,
                              void* d_out, int out_size, void* d_ws, size_t ws_size,
                              hipStream_t stream) {
    (void)in_sizes; (void)n_in; (void)out_size; (void)ws_size;
    uint8_t* ws = (uint8_t*)d_ws;
    size_t off = 0;
    auto alloc = [&](size_t bytes) -> uint8_t* {
        uint8_t* p = ws + off;
        off = (off + bytes + 255) & ~(size_t)255;
        return p;
    };

    f16*   Xn    = (f16*)  alloc((size_t)M0 * 256 * 2);   // also FC1 out (same size)
    float* X     = (float*)alloc((size_t)M0 * 64 * 4);    // residual stream (block0)
    f16*   Aw    = (f16*)  alloc((size_t)M0 * 64 * 2);    // LN'd windowed tokens / tr pack
    f16*   Qb    = (f16*)  alloc((size_t)M0 * 64 * 2);
    f16*   Kb    = (f16*)  alloc((size_t)M0 * 64 * 2);
    f16*   Vb    = (f16*)  alloc((size_t)M0 * 64 * 2);
    f16*   Attn  = (f16*)  alloc((size_t)M0 * 64 * 2);    // also final prelu pack
    f16*   Hln   = (f16*)  alloc((size_t)M0 * 64 * 2);
    float* X1    = (float*)alloc((size_t)M1 * 64 * 4);    // residual stream (block1)
    float* OutTr = (float*)alloc((size_t)M0 * 192 * 4);
    f16*   Aca   = (f16*)  alloc((size_t)M0 * 192 * 2);   // also A_pw
    float* QW    = (float*)alloc((size_t)M0 * 256 * 4);   // also P1 / C1
    float* KW    = (float*)alloc((size_t)M0 * 256 * 4);   // also OA2
    float* VW    = (float*)alloc((size_t)M0 * 192 * 4);   // also Acc (f16 M0*384)
    f16* wqkv = (f16*)alloc(64 * 192 * 2);
    f16* wproj = (f16*)alloc(64 * 64 * 2);
    f16* wfc1 = (f16*)alloc(64 * 256 * 2);
    f16* wfc2 = (f16*)alloc(256 * 64 * 2);
    f16* wbn  = (f16*)alloc(256 * 64 * 2);
    f16* wout = (f16*)alloc(64 * 256 * 2);
    f16* wtr  = (f16*)alloc(64 * 192 * 2);
    f16* wpw  = (f16*)alloc(192 * 192 * 2);
    f16* wqca = (f16*)alloc(192 * 256 * 2);
    f16* wkca = (f16*)alloc(192 * 256 * 2);
    f16* wvca = (f16*)alloc(192 * 192 * 2);
    f16* wcc  = (f16*)alloc(384 * 64 * 2);
    float* sQb = (float*)alloc(256 * 4);
    float* sKb = (float*)alloc(256 * 4);
    float* sVb = (float*)alloc(192 * 4);
    float* sQg = (float*)alloc(256 * 4); float* sQbe = (float*)alloc(256 * 4);
    float* sKg = (float*)alloc(256 * 4); float* sKbe = (float*)alloc(256 * 4);
    float* sVg = (float*)alloc(192 * 4); float* sVbe = (float*)alloc(192 * 4);
    float* logits = (float*)alloc(32 * 4);
    float* wts    = (float*)alloc(32 * 4);

    // d_in layout (dict insertion order): 0=x, 1=cln_g, 2=cln_b, 3=bn_w, 4=out_a,
    // 5=out_w, 6=out_b, 7..58 block0 layers (4x13), 59.. block0.tac, 120..171 block1 layers.
    const float* xin = (const float*)d_in[0];

    // 1) channelwise LN + bottleneck 256->64
    k_cln<<<(M0 + 255) / 256, 256, 0, stream>>>(xin, (const float*)d_in[1], (const float*)d_in[2], Xn);
    k_conv_w_t<<<(256 * 64 + 255) / 256, 256, 0, stream>>>((const float*)d_in[3], wbn, 256, 64, 64, 0);
    launch_gemm(stream, Xn, wbn, nullptr, M0, 64, 256, E_F32,
                X, nullptr, nullptr, nullptr, nullptr, nullptr, nullptr, 0, 0);

    // 2) block0: 4 swin blocks (alternating shift)
    for (int blk = 0; blk < 4; ++blk)
        run_swin(stream, d_in, 7 + blk * 13, 4, blk & 1, X,
                 Aw, Qb, Kb, Vb, Attn, Hln, (f16*)Xn, wqkv, wproj, wfc1, wfc2);

    // 3) TAC channel attention
    {
        k_conv_w_t<<<(64 * 192 + 255) / 256, 256, 0, stream>>>((const float*)d_in[59], wtr, 64, 192, 192, 0);
        k_pack_tac<<<(M0 * 64 + 255) / 256, 256, 0, stream>>>(X, Aw);
        launch_gemm(stream, Aw, wtr, (const float*)d_in[60], M0, 192, 64, E_PRELU,
                    OutTr, nullptr, nullptr, (const float*)d_in[61], nullptr, nullptr, nullptr, 0, 0);
        k_pack_ca<<<(M0 * 192 + 255) / 256, 256, 0, stream>>>(OutTr, Aca);
        for (int h = 0; h < 4; ++h) {
            k_conv_w_t<<<(192 * 64 + 255) / 256, 256, 0, stream>>>((const float*)d_in[62 + h * 4 + 0], wqca, 192, 64, 256, h * 64);
            k_copy_f32<<<1, 64, 0, stream>>>(sQb + h * 64, (const float*)d_in[62 + h * 4 + 1], 64);
            k_copy_f32<<<1, 64, 0, stream>>>(sQg + h * 64, (const float*)d_in[62 + h * 4 + 2], 64);
            k_copy_f32<<<1, 64, 0, stream>>>(sQbe + h * 64, (const float*)d_in[62 + h * 4 + 3], 64);
            k_conv_w_t<<<(192 * 64 + 255) / 256, 256, 0, stream>>>((const float*)d_in[78 + h * 4 + 0], wkca, 192, 64, 256, h * 64);
            k_copy_f32<<<1, 64, 0, stream>>>(sKb + h * 64, (const float*)d_in[78 + h * 4 + 1], 64);
            k_copy_f32<<<1, 64, 0, stream>>>(sKg + h * 64, (const float*)d_in[78 + h * 4 + 2], 64);
            k_copy_f32<<<1, 64, 0, stream>>>(sKbe + h * 64, (const float*)d_in[78 + h * 4 + 3], 64);
            k_conv_w_t<<<(192 * 48 + 255) / 256, 256, 0, stream>>>((const float*)d_in[94 + h * 4 + 0], wvca, 192, 48, 192, h * 48);
            k_copy_f32<<<1, 64, 0, stream>>>(sVb + h * 48, (const float*)d_in[94 + h * 4 + 1], 48);
            k_copy_f32<<<1, 64, 0, stream>>>(sVg + h * 48, (const float*)d_in[94 + h * 4 + 2], 48);
            k_copy_f32<<<1, 64, 0, stream>>>(sVbe + h * 48, (const float*)d_in[94 + h * 4 + 3], 48);
        }
        launch_gemm(stream, Aca, wqca, sQb, M0, 256, 192, E_RELU, QW, nullptr, nullptr, nullptr, nullptr, nullptr, nullptr, 0, 0);
        launch_gemm(stream, Aca, wkca, sKb, M0, 256, 192, E_RELU, KW, nullptr, nullptr, nullptr, nullptr, nullptr, nullptr, 0, 0);
        launch_gemm(stream, Aca, wvca, sVb, M0, 192, 192, E_RELU, VW, nullptr, nullptr, nullptr, nullptr, nullptr, nullptr, 0, 0);
        k_chunk_ln<<<(M0 * 4 + 255) / 256, 256, 0, stream>>>(QW, sQg, sQbe, M0, 256, 64);
        k_chunk_ln<<<(M0 * 4 + 255) / 256, 256, 0, stream>>>(KW, sKg, sKbe, M0, 256, 64);
        k_chunk_ln<<<(M0 * 4 + 255) / 256, 256, 0, stream>>>(VW, sVg, sVbe, M0, 192, 48);
        k_ca_dots<<<32, 256, 0, stream>>>(QW, KW, logits);
        k_ca_soft<<<1, 32, 0, stream>>>(logits, wts);
        k_ca_mix<<<(M0 * 192 + 255) / 256, 256, 0, stream>>>(VW, wts, Aca);   // Aca = A_pw now
        k_conv_w_t<<<(192 * 192 + 255) / 256, 256, 0, stream>>>((const float*)d_in[110], wpw, 192, 192, 192, 0);
        float* P1 = QW; float* OA2 = KW;
        launch_gemm(stream, Aca, wpw, (const float*)d_in[111], M0, 192, 192, E_RELU,
                    P1, nullptr, nullptr, nullptr, nullptr, nullptr, nullptr, 0, 0);
        k_ln_res_prelu<<<(M0 + 255) / 256, 256, 0, stream>>>(P1, OutTr, (const float*)d_in[112],
                                                             (const float*)d_in[113], (const float*)d_in[114], OA2);
        f16* Acc = (f16*)VW;
        k_pack_cc<<<(M0 * 384 + 255) / 256, 256, 0, stream>>>(OutTr, OA2, Acc);
        k_conv_w_t<<<(384 * 64 + 255) / 256, 256, 0, stream>>>((const float*)d_in[115], wcc, 384, 64, 64, 0);
        float* C1 = QW;
        launch_gemm(stream, Acc, wcc, (const float*)d_in[116], M0, 64, 384, E_PRELU,
                    C1, nullptr, nullptr, (const float*)d_in[117], nullptr, nullptr, nullptr, 0, 0);
        k_cc_ln_res<<<(M0 + 255) / 256, 256, 0, stream>>>(C1, (const float*)d_in[118], (const float*)d_in[119], X);
    }

    // 4) block1: mean over mic channels, then 4 swin blocks on 2 images
    k_mean_ch<<<(M1 * 64 + 255) / 256, 256, 0, stream>>>(X, X1);
    for (int blk = 0; blk < 4; ++blk)
        run_swin(stream, d_in, 120 + blk * 13, 2, blk & 1, X1,
                 Aw, Qb, Kb, Vb, Attn, Hln, (f16*)Xn, wqkv, wproj, wfc1, wfc2);

    // 5) prelu + 64->256 output projection into d_out (B,256,F,T)
    k_prelu_pack<<<(M1 * 64 + 255) / 256, 256, 0, stream>>>(X1, (const float*)d_in[4], Attn);
    k_conv_w_t<<<(64 * 256 + 255) / 256, 256, 0, stream>>>((const float*)d_in[5], wout, 64, 256, 256, 0);
    launch_gemm(stream, Attn, wout, (const float*)d_in[6], M1, 256, 64, E_OUT,
                (float*)d_out, nullptr, nullptr, nullptr, nullptr, nullptr, nullptr, 0, LQ);
}
